// GraphAttentionLayer_20048907337750
// MI455X (gfx1250) — matrix operand order repaired and checked by reading, compiled, not run on hardware
//
#include <hip/hip_runtime.h>
#include <hip/hip_bf16.h>

#define ALPHA   0.2f
#define NEG_INF (-9.0e15f)

#define Bsz  8
#define Nn   2048
#define Ff   128
#define BN   (Bsz * Nn)   // 16384

typedef float        v2f  __attribute__((ext_vector_type(2)));
typedef float        v8f  __attribute__((ext_vector_type(8)));
typedef _Float16     v8h  __attribute__((ext_vector_type(8)));
typedef _Float16     v16h __attribute__((ext_vector_type(16)));
typedef unsigned int v4u  __attribute__((ext_vector_type(4)));
typedef int          v4i  __attribute__((ext_vector_type(4)));
typedef int          v8i  __attribute__((ext_vector_type(8)));

union V16U { v16h v; v8h h[2]; };

// ---------------------------------------------------------------------------
// Kernel 1: h[r,o] = sum_f x[r,f] * W[f,o] + bW[o]   (16384 rows)
// fp32 WMMA 16x16x4; also emits hT16[b][o][j] = (f16)h — the pre-transposed
// f16 operand the attention GEMM's TDM loads consume (one v8h store/lane).
// ---------------------------------------------------------------------------
__global__ __launch_bounds__(256) void gat_fc(const float* __restrict__ x,
                                              const float* __restrict__ W,
                                              const float* __restrict__ bW,
                                              float* __restrict__ h,
                                              _Float16* __restrict__ hT16) {
    const int rowbase = blockIdx.x * 16;
    const int t = threadIdx.x, lane = t & 31, wave = t >> 5;

    __shared__ float xs[16][132];   // 132 mod 64 = 4 -> conflict-free A reads
    {
        int k = t & 127, rr = t >> 7;
        const float* xp = x + (size_t)rowbase * Ff + k;
        for (int r = 0; r < 8; ++r) {
            int row = rr + r * 2;
            xs[row][k] = xp[(size_t)row * Ff];
        }
    }
    __syncthreads();

    const int n0   = wave * 16;
    const int ln16 = lane & 15;
    const int koff = (lane < 16) ? 0 : 2;

    v8f c = {0.f, 0.f, 0.f, 0.f, 0.f, 0.f, 0.f, 0.f};
    for (int k = 0; k < Ff; k += 4) {
        v2f a, b;
        a.x = xs[ln16][k + koff];
        a.y = xs[ln16][k + koff + 1];
        b.x = W[(size_t)(k + koff)     * Ff + n0 + ln16];
        b.y = W[(size_t)(k + koff + 1) * Ff + n0 + ln16];
        c = __builtin_amdgcn_wmma_f32_16x16x4_f32(false, a, false, b,
                                                  (short)0, c, false, false);
    }
    const float bb = bW[n0 + ln16];
    union { v8h v; _Float16 e[8]; } pk;
    for (int v = 0; v < 8; ++v) {
        int m = v + ((lane < 16) ? 0 : 8);
        float r = c[v] + bb;
        h[((size_t)rowbase + m) * Ff + n0 + ln16] = r;
        pk.e[v] = (_Float16)r;                    // m=v..v+7 -> consecutive j
    }
    const int brow = rowbase >> 11;               // batch (16 | 2048)
    const int jb   = (rowbase & 2047) + ((lane < 16) ? 0 : 8);
    *(v8h*)(hT16 + ((size_t)(brow * Ff + n0 + ln16)) * Nn + jb) = pk.v;
}

// ---------------------------------------------------------------------------
// Kernel 2: s1[r] = h[r,:].a1 ; s2[r] = h[r,:].a2   (wave32 per row)
// ---------------------------------------------------------------------------
__global__ __launch_bounds__(256) void gat_scores(const float* __restrict__ h,
                                                  const float* __restrict__ a1,
                                                  const float* __restrict__ a2,
                                                  float* __restrict__ s1,
                                                  float* __restrict__ s2) {
    const int t = threadIdx.x, lane = t & 31, wave = t >> 5;
    const int row = blockIdx.x * 8 + wave;
    const float* hp = h + (size_t)row * Ff;
    float p1 = 0.f, p2 = 0.f;
    for (int e = 0; e < 4; ++e) {
        float hv = hp[lane + 32 * e];
        p1 += hv * a1[lane + 32 * e];
        p2 += hv * a2[lane + 32 * e];
    }
    for (int off = 16; off > 0; off >>= 1) {
        p1 += __shfl_down(p1, off, 32);
        p2 += __shfl_down(p2, off, 32);
    }
    if (lane == 0) { s1[row] = p1; s2[row] = p2; }
}

// ---------------------------------------------------------------------------
// Kernel 3: single coalesced pass over adj (the 134 MB HBM floor):
//  - online column max / rescaled sum  (softmax over axis=1, i.e. over i)
//  - packs adj>0 into a bitmask maskJ[b][i/32][j] (4 MB) so the GEMM pass
//    never re-reads adj (32x pass-2 traffic cut).
// ---------------------------------------------------------------------------
__global__ __launch_bounds__(256) void gat_colstats(const int* __restrict__ adj,
                                                    const float* __restrict__ s1,
                                                    const float* __restrict__ s2,
                                                    const float* __restrict__ bap,
                                                    float* __restrict__ colmax,
                                                    float* __restrict__ rdenom,
                                                    unsigned* __restrict__ maskJ) {
    const int b = blockIdx.x >> 3;
    const int j = (blockIdx.x & 7) * 256 + threadIdx.x;
    const float b_a = *bap;
    const float s2v = s2[b * Nn + j] + b_a;
    const int* ap    = adj + (size_t)b * Nn * Nn + j;
    const float* s1p = s1 + b * Nn;

    float m = -INFINITY, d = 0.f;
    unsigned wacc = 0;
    for (int i = 0; i < Nn; ++i) {
        if (i + 8 < Nn)
            __builtin_prefetch(ap + (size_t)(i + 8) * Nn, 0, 3);  // global_prefetch_b8
        int   a  = ap[(size_t)i * Nn];
        wacc |= (a > 0 ? 1u : 0u) << (i & 31);
        float sc = s1p[i] + s2v;
        sc = (sc > 0.f) ? sc : ALPHA * sc;
        float v  = (a > 0) ? sc : NEG_INF;
        float nm = fmaxf(m, v);
        d = d * __expf(m - nm) + __expf(v - nm);
        m = nm;
        if ((i & 31) == 31) {                      // coalesced mask store
            maskJ[((size_t)b * 64 + (i >> 5)) * Nn + j] = wacc;
            wacc = 0;
        }
    }
    colmax[b * Nn + j] = m;
    rdenom[b * Nn + j] = 1.0f / d;
}

// ---------------------------------------------------------------------------
// Kernel 4: h'[b,i,:] = relu( sum_j P[b,i,j] * h[b,j,:] )
// Per block: 64 rows x 128 cols, one b. Per 64-j chunk:
//   wave 0 launches a TDM tensor_load_to_lds of the 128x64 f16 hT16 tile
//   (pad_enable: 32-DWORD rows + 4-DWORD pad == the 72-half LDS pitch),
//   all 256 threads build the softmaxed f16 P tile from the bitmask while
//   the DMA flies, then s_wait_tensorcnt + barrier, then f16 WMMA 16x16x32
//   with persistent f32 accumulators.
// ---------------------------------------------------------------------------
__global__ __launch_bounds__(256) void gat_attn_gemm(const _Float16* __restrict__ hT16,
                                                     const unsigned* __restrict__ maskJ,
                                                     const float* __restrict__ s1,
                                                     const float* __restrict__ s2,
                                                     const float* __restrict__ colmax,
                                                     const float* __restrict__ rdenom,
                                                     const float* __restrict__ bap,
                                                     float* __restrict__ out) {
    const int blk  = blockIdx.x;          // 256 blocks = 8 b * 32 row-slabs
    const int b    = blk >> 5;
    const int i0   = (blk & 31) * 64;
    const int t    = threadIdx.x, lane = t & 31, wave = t >> 5;
    const float b_a = *bap;

    __shared__ _Float16 p16[64][72];      // P tile   [i][j]
    __shared__ _Float16 hsT[128][72];     // h chunk  [o][j]  (filled by TDM)
    __shared__ float    ls1[64];

    if (t < 64) ls1[t] = s1[b * Nn + i0 + t];

    const int mrow  = (wave & 3) * 16;
    const int cbase = (wave >> 2) * 64;
    const int ln16  = lane & 15;
    const int kA    = (lane < 16) ? 0 : 8;    // A (f16 16x32) K base
    const int kB    = (lane < 16) ? 0 : 8;    // B (f16 32x16) K base

    const unsigned lds_hsT = (unsigned)(size_t)(&hsT[0][0]);   // LDS byte offset

    v8f acc[4];
    for (int ct = 0; ct < 4; ++ct)
        acc[ct] = (v8f){0.f, 0.f, 0.f, 0.f, 0.f, 0.f, 0.f, 0.f};

    __syncthreads();

    for (int j0 = 0; j0 < Nn; j0 += 64) {
        // ---- wave 0: kick TDM load of hT16[b][0..127][j0..j0+63] -> hsT ----
        if (wave == 0) {
            size_t ga = (size_t)(hT16 + (size_t)b * Ff * Nn + j0);  // byte addr
            v4u g0 = { 1u,                                  // count=1 (valid D#)
                       lds_hsT,                             // lds_addr
                       (unsigned)(ga & 0xFFFFFFFFu),
                       (unsigned)((ga >> 32) & 0x01FFFFFFu) | 0x80000000u }; // type=2
            v8i g1 = { (int)((1u << 16) |                   // data_size = 2B
                             (1u << 20) |                   // pad_enable
                             (4u << 22) |                   // pad_interval: 32 DWORDs
                             (3u << 25)),                   // pad_amount:   4 DWORDs
                       (int)(2048u << 16),                  // tensor_dim0 = 2048 (j)
                       (int)(128u  << 16),                  // tensor_dim1 = 128  (o)
                       (int)(64u   << 16),                  // tile_dim0   = 64
                       128,                                 // tile_dim1   = 128
                       2048,                                // tensor_dim0_stride
                       0, 4 };                              // tensor_dim1_stride = 2048*128
            v4i gz4 = {0, 0, 0, 0};
            v8i gz8 = {0, 0, 0, 0, 0, 0, 0, 0};
            __builtin_amdgcn_tensor_load_to_lds(g0, g1, gz4, gz4, gz8, 0);
        }

        // ---- build softmaxed P tile from bitmask (overlaps the DMA) ----
        {
            const int   jj    = t & 63;
            const int   jglob = j0 + jj;
            const float s2v   = s2[b * Nn + jglob] + b_a;
            const float cm    = colmax[b * Nn + jglob];
            const float rd    = rdenom[b * Nn + jglob];
            const unsigned* mp = maskJ + ((size_t)b * 64 + (i0 >> 5)) * Nn + jglob;
            const unsigned w0 = mp[0];
            const unsigned w1 = mp[Nn];
            for (int r = 0; r < 16; ++r) {
                int ii = (t >> 6) + (r << 2);
                unsigned bit = ((ii < 32) ? (w0 >> ii) : (w1 >> (ii - 32))) & 1u;
                float sc = ls1[ii] + s2v;
                sc = (sc > 0.f) ? sc : ALPHA * sc;
                float val = bit ? sc : NEG_INF;
                p16[ii][jj] = (_Float16)(__expf(val - cm) * rd);
            }
        }

        if (wave == 0) __builtin_amdgcn_s_wait_tensorcnt(0);
        __syncthreads();

        // ---- f16 WMMA over the 64-deep j chunk (two K=32 steps) ----
        for (int kc = 0; kc < 64; kc += 32) {
            V16U A;
            A.h[0] = *(const v8h*)&p16[mrow + ln16][kc + kA];
            A.h[1] = *(const v8h*)&p16[mrow + ln16][kc + kA + 16];
            for (int ct = 0; ct < 4; ++ct) {
                int n = cbase + ct * 16 + ln16;
                V16U Bm;
                Bm.h[0] = *(const v8h*)&hsT[n][kc + kB];
                Bm.h[1] = *(const v8h*)&hsT[n][kc + kB + 16];
                acc[ct] = __builtin_amdgcn_wmma_f32_16x16x32_f16(
                    false, A.v, false, Bm.v, (short)0, acc[ct], false, false);
            }
        }
        __syncthreads();
    }

    // ---- ReLU + store ----
    for (int ct = 0; ct < 4; ++ct) {
        int o = cbase + ct * 16 + ln16;
        for (int v = 0; v < 8; ++v) {
            int   m = v + ((lane < 16) ? 0 : 8);
            float r = acc[ct][v];
            out[((size_t)b * Nn + i0 + mrow + m) * Ff + o] = (r > 0.f) ? r : 0.f;
        }
    }
}

// ---------------------------------------------------------------------------
extern "C" void kernel_launch(void* const* d_in, const int* in_sizes, int n_in,
                              void* d_out, int out_size, void* d_ws, size_t ws_size,
                              hipStream_t stream) {
    const float* x   = (const float*)d_in[0];
    const int*   adj = (const int*)  d_in[1];
    const float* W   = (const float*)d_in[2];
    const float* bW  = (const float*)d_in[3];
    const float* a1  = (const float*)d_in[4];
    const float* a2  = (const float*)d_in[5];
    const float* b_a = (const float*)d_in[6];   // scalar (1 element)

    float*    h     = (float*)d_ws;                              // 8 MB
    _Float16* hT16  = (_Float16*)(h + (size_t)BN * Ff);          // 4 MB
    unsigned* maskJ = (unsigned*)(hT16 + (size_t)BN * Ff);       // 4 MB
    float*    s1    = (float*)(maskJ + (size_t)Bsz * 64 * Nn);
    float*    s2    = s1   + BN;
    float*    cmax  = s2   + BN;
    float*    rden  = cmax + BN;

    hipLaunchKernelGGL(gat_fc,        dim3(BN / 16), dim3(256), 0, stream,
                       x, W, bW, h, hT16);
    hipLaunchKernelGGL(gat_scores,    dim3(BN / 8),  dim3(256), 0, stream,
                       h, a1, a2, s1, s2);
    hipLaunchKernelGGL(gat_colstats,  dim3(Bsz * Nn / 256), dim3(256), 0, stream,
                       adj, s1, s2, b_a, cmax, rden, maskJ);
    hipLaunchKernelGGL(gat_attn_gemm, dim3(Bsz * (Nn / 64)), dim3(256), 0, stream,
                       hT16, maskJ, s1, s2, cmax, rden, b_a, (float*)d_out);
}
